// MultiScaleTemporal_63599875719568
// MI455X (gfx1250) — matrix operand order repaired and checked by reading, not compiled, not run on hardware
//
#include <hip/hip_runtime.h>
#include <hip/hip_bf16.h>

#define D_MODEL 256
#define D_INNER 512
#define D_STATE 16
#define D_CONV  4
#define DT_RANK 16
#define NS      3
#define BB      8
#define TT      48
#define NNP     207
#define LL      12
#define BN      (BB*NNP)      // 1656 sequences per scale
#define ROWS    (BN*LL)       // 19872 token rows per scale (divisible by 16)
#define MPAD    1664          // BN padded to multiple of 16 for out_proj GEMM
#define XP_N    64            // x_proj output channels padded 48 -> 64

typedef _Float16 v8h  __attribute__((ext_vector_type(8)));
typedef _Float16 v16h __attribute__((ext_vector_type(16)));
typedef float    v8f  __attribute__((ext_vector_type(8)));

// ---------------------------------------------------------------------------
// f32 -> f16 convert (for weight matrices)
// ---------------------------------------------------------------------------
__global__ __launch_bounds__(256) void f32_to_f16_kernel(
    const float* __restrict__ src, _Float16* __restrict__ dst, long n)
{
    long g = (long)blockIdx.x * 256 + threadIdx.x;
    if (g < n) dst[g] = (_Float16)src[g];
}

// ---------------------------------------------------------------------------
// x_proj_w (NS,48,512) f32 -> (NS,64,512) f16 with zero-padded rows 48..63
// ---------------------------------------------------------------------------
__global__ __launch_bounds__(256) void convert_xproj_kernel(
    const float* __restrict__ src, _Float16* __restrict__ dst)
{
    int g = blockIdx.x * 256 + threadIdx.x;       // NS*64*512 = 98304
    if (g >= NS * XP_N * D_INNER) return;
    int d = g % D_INNER;
    int e = (g / D_INNER) % XP_N;
    int s = g / (D_INNER * XP_N);
    dst[g] = (e < 48) ? (_Float16)src[((long)s * 48 + e) * D_INNER + d]
                      : (_Float16)0.f;
}

// ---------------------------------------------------------------------------
// Gather the 3 temporal views of x into (NS, ROWS, D_MODEL) f16, row order
// row = ((b*N + n)*12 + l) matching the reference reshape.
// ---------------------------------------------------------------------------
__global__ __launch_bounds__(256) void gather_x_kernel(
    const float* __restrict__ x, _Float16* __restrict__ xh)
{
    long g = (long)blockIdx.x * 256 + threadIdx.x;
    if (g >= (long)NS * ROWS * D_MODEL) return;
    int  d  = (int)(g & (D_MODEL - 1));
    long r  = g >> 8;                 // token row index over NS*ROWS
    int  l  = (int)(r % LL);
    long pr = r / LL;
    int  p  = (int)(pr % BN);
    int  s  = (int)(pr / BN);
    int  b  = p / NNP, n = p % NNP;
    int  t  = (s == 0) ? (36 + l) : (s == 1) ? (24 + 2 * l) : (4 * l);
    xh[g] = (_Float16)x[(((long)b * TT + t) * NNP + n) * D_MODEL + d];
}

// ---------------------------------------------------------------------------
// WMMA GEMM: C[m,n] = sum_k A[m,k] * W[n,k]  (A: MxK f16 row-major,
// W: NxK f16 row-major, C = A * W^T).  One wave computes a 16x64 strip
// (4 accumulators; A fragment reused across 4 B tiles -> 4x lower A traffic
// and 4 independent WMMA chains).  4 waves/block cover 4 consecutive m-tiles
// sharing the same B tiles (L0/L2 hits).  N must be a multiple of 64.
// ---------------------------------------------------------------------------
__global__ __launch_bounds__(128) void wmma_gemm_kernel(
    const _Float16* __restrict__ A, const _Float16* __restrict__ W,
    float* __restrict__ Cf, _Float16* __restrict__ Ch,
    int N, int K, int Mrows, int Mvalid, long sA, long sB, long sC)
{
    const int lane = threadIdx.x;        // 0..31
    const int wave = threadIdx.y;        // 0..3
    const int mt   = blockIdx.x * 4 + wave;
    if (mt * 16 >= Mrows) return;        // wave-uniform exit
    const int nb   = blockIdx.y * 4;     // first of 4 n-tiles
    const int sc   = blockIdx.z;
    const int half = lane >> 4;          // 0 | 1
    const int lid  = lane & 15;
    const int ak   = half * 8;           // A frag: lanes16-31 at K+8
    const int bk   = half * 8;           // B frag: lanes16-31 at K+8

    const _Float16* Ap = A + sc * sA + (long)(mt * 16 + lid) * K;
    const _Float16* Bp[4];
#pragma unroll
    for (int j = 0; j < 4; j++)
        Bp[j] = W + sc * sB + (long)((nb + j) * 16 + lid) * K;

    v8f acc[4];
#pragma unroll
    for (int j = 0; j < 4; j++)
#pragma unroll
        for (int i = 0; i < 8; i++) acc[j][i] = 0.0f;

    for (int kb = 0; kb < K; kb += 32) {
        union { v16h v; v8h h[2]; } a, b[4];
        a.h[0] = *(const v8h*)(Ap + kb + ak);
        a.h[1] = *(const v8h*)(Ap + kb + 16 + ak);
#pragma unroll
        for (int j = 0; j < 4; j++) {
            b[j].h[0] = *(const v8h*)(Bp[j] + kb + bk);
            b[j].h[1] = *(const v8h*)(Bp[j] + kb + 16 + bk);
        }
#pragma unroll
        for (int j = 0; j < 4; j++)
            acc[j] = __builtin_amdgcn_wmma_f32_16x16x32_f16(
                false, a.v, false, b[j].v, (short)0, acc[j], false, false);
    }

    const int rbase = mt * 16 + half * 8;
#pragma unroll
    for (int j = 0; j < 4; j++) {
        const int col = (nb + j) * 16 + lid;
#pragma unroll
        for (int r = 0; r < 8; r++) {
            const int row = rbase + r;
            if (row < Mvalid) {
                if (Ch) Ch[sc * sC + (long)row * N + col] = (_Float16)acc[j][r];
                else    Cf[sc * sC + (long)row * N + col] = acc[j][r];
            }
        }
    }
}

// ---------------------------------------------------------------------------
// Depthwise causal conv (4 taps) + SiLU over xr = xz[:, :, 0:512]
// ---------------------------------------------------------------------------
__global__ __launch_bounds__(256) void conv_silu_kernel(
    const _Float16* __restrict__ xz,   // (NS, ROWS, 1024)
    const float* __restrict__ cw,      // (NS, 512, 4)
    const float* __restrict__ cb,      // (NS, 512)
    _Float16* __restrict__ xc)         // (NS, ROWS, 512)
{
    long g = (long)blockIdx.x * 256 + threadIdx.x;
    if (g >= (long)NS * ROWS * D_INNER) return;
    int  d = (int)(g % D_INNER);
    long r = g / D_INNER;               // global token row over NS*ROWS
    int  l = (int)(r % LL);
    long s = r / ROWS;
    long base = r - l;                  // first row of this sequence
    float v = cb[s * D_INNER + d];
#pragma unroll
    for (int k = 0; k < D_CONV; k++) {
        int li = l - (D_CONV - 1) + k;
        if (li >= 0)
            v += (float)xz[(base + li) * (2 * D_INNER) + d] *
                 cw[(s * D_INNER + d) * D_CONV + k];
    }
    v = v / (1.f + __expf(-v));         // SiLU
    xc[g] = (_Float16)v;
}

// ---------------------------------------------------------------------------
// dt projection + softplus + selective scan + gate.  One block per (bn, s),
// one thread per inner channel d; 16 states in registers; dtr/B/C in LDS.
// Writes gated y (f16) into the M-padded buffer for the out_proj GEMM.
// ---------------------------------------------------------------------------
__global__ __launch_bounds__(512) void scan_kernel(
    const _Float16* __restrict__ xc,    // (NS, ROWS, 512)
    const _Float16* __restrict__ xz,    // (NS, ROWS, 1024)  (z = [:,512:])
    const float* __restrict__ xdbl,     // (NS, ROWS, 64)  cols 0..47 valid
    const float* __restrict__ dtW,      // (NS, 512, 16)
    const float* __restrict__ dtb,      // (NS, 512)
    const float* __restrict__ Alog,     // (NS, 512, 16)
    const float* __restrict__ Dp,       // (NS, 512)
    _Float16* __restrict__ yh)          // (NS, MPAD, 512)
{
    const int bn = blockIdx.x;
    const int s  = blockIdx.y;
    const int d  = threadIdx.x;
    if (bn >= BN) {                     // zero the GEMM padding rows
        yh[((long)s * MPAD + bn) * D_INNER + d] = (_Float16)0.f;
        return;
    }
    __shared__ float lds[LL * 48];      // dtr(16) | B(16) | C(16) per step
    const long rb = (long)s * ROWS + (long)bn * LL;
    for (int i = d; i < LL * 48; i += D_INNER)
        lds[i] = xdbl[(rb + i / 48) * XP_N + (i % 48)];
    __syncthreads();

    float w[DT_RANK], A[D_STATE], h[D_STATE];
#pragma unroll
    for (int r = 0; r < DT_RANK; r++) w[r] = dtW[((long)s * D_INNER + d) * DT_RANK + r];
#pragma unroll
    for (int n = 0; n < D_STATE; n++) A[n] = -__expf(Alog[((long)s * D_INNER + d) * D_STATE + n]);
#pragma unroll
    for (int n = 0; n < D_STATE; n++) h[n] = 0.f;
    const float bias = dtb[s * D_INNER + d];

    float xt = 0.f;
    for (int l = 0; l < LL; l++) {
        xt = (float)xc[(rb + l) * D_INNER + d];
        float acc = bias;
#pragma unroll
        for (int r = 0; r < DT_RANK; r++) acc += lds[l * 48 + r] * w[r];
        float dt = (acc > 20.f) ? acc : log1pf(__expf(acc));   // softplus
#pragma unroll
        for (int n = 0; n < D_STATE; n++)
            h[n] = __expf(dt * A[n]) * h[n] + dt * lds[l * 48 + 16 + n] * xt;
    }
    float y = 0.f;
#pragma unroll
    for (int n = 0; n < D_STATE; n++) y += h[n] * lds[(LL - 1) * 48 + 32 + n];
    y += Dp[s * D_INNER + d] * xt;
    float z = (float)xz[(rb + LL - 1) * (2 * D_INNER) + D_INNER + d];
    y *= z / (1.f + __expf(-z));        // y * silu(z_last)
    yh[((long)s * MPAD + bn) * D_INNER + d] = (_Float16)y;
}

// ---------------------------------------------------------------------------
// Per-position attention over the 3 scales: score_s = feats_s . attn_w + b,
// softmax over s, weighted sum.  One block (256 threads) per (b,n).
// ---------------------------------------------------------------------------
__global__ __launch_bounds__(256) void attn_out_kernel(
    const float* __restrict__ feats,   // (NS, BN, 256)
    const float* __restrict__ attn_w,  // (256)
    const float* __restrict__ attn_b,  // (1)
    float* __restrict__ out)           // (BN, 256)
{
    const int p = blockIdx.x;
    const int d = threadIdx.x;
    __shared__ float red[256];
    float f[NS], sc[NS];
#pragma unroll
    for (int s = 0; s < NS; s++) {
        f[s] = feats[((long)s * BN + p) * 256 + d];
        red[d] = f[s] * attn_w[d];
        __syncthreads();
        for (int st = 128; st > 0; st >>= 1) {
            if (d < st) red[d] += red[d + st];
            __syncthreads();
        }
        sc[s] = red[0] + attn_b[0];
        __syncthreads();
    }
    float m  = fmaxf(sc[0], fmaxf(sc[1], sc[2]));
    float e0 = __expf(sc[0] - m), e1 = __expf(sc[1] - m), e2 = __expf(sc[2] - m);
    float inv = 1.f / (e0 + e1 + e2);
    out[(long)p * 256 + d] = (f[0] * e0 + f[1] * e1 + f[2] * e2) * inv;
}

// ---------------------------------------------------------------------------
extern "C" void kernel_launch(void* const* d_in, const int* in_sizes, int n_in,
                              void* d_out, int out_size, void* d_ws, size_t ws_size,
                              hipStream_t stream)
{
    const float* x         = (const float*)d_in[0];
    const float* in_proj_w = (const float*)d_in[1];   // (3,1024,256)
    const float* conv_w    = (const float*)d_in[2];   // (3,512,4)
    const float* conv_b    = (const float*)d_in[3];   // (3,512)
    const float* x_proj_w  = (const float*)d_in[4];   // (3,48,512)
    const float* dt_proj_w = (const float*)d_in[5];   // (3,512,16)
    const float* dt_proj_b = (const float*)d_in[6];   // (3,512)
    const float* A_log     = (const float*)d_in[7];   // (3,512,16)
    const float* Dp        = (const float*)d_in[8];   // (3,512)
    const float* out_proj  = (const float*)d_in[9];   // (3,256,512)
    const float* attn_w    = (const float*)d_in[10];  // (1,256)
    const float* attn_b    = (const float*)d_in[11];  // (1)
    float* out = (float*)d_out;

    // workspace layout (all offsets multiples of 256 B)
    char* w = (char*)d_ws;
    _Float16* xh     = (_Float16*)w;  w += (long)NS * ROWS * D_MODEL * 2;   // 30.5 MB
    _Float16* wh_in  = (_Float16*)w;  w += (long)NS * 1024 * 256 * 2;       // 1.6 MB
    _Float16* wh_xp  = (_Float16*)w;  w += (long)NS * XP_N * 512 * 2;       // 0.2 MB
    _Float16* wh_out = (_Float16*)w;  w += (long)NS * 256 * 512 * 2;        // 0.8 MB
    _Float16* xz     = (_Float16*)w;  w += (long)NS * ROWS * 1024 * 2;      // 122 MB
    _Float16* xc     = (_Float16*)w;  w += (long)NS * ROWS * 512 * 2;       // 61 MB
    float*    xdbl   = (float*)w;     w += (long)NS * ROWS * XP_N * 4;      // 15.3 MB
    _Float16* yh     = (_Float16*)w;  w += (long)NS * MPAD * 512 * 2;       // 5.1 MB
    float*    feats  = (float*)w;     w += (long)NS * BN * 256 * 4;         // 5.1 MB

    const int MT  = ROWS / 16;                 // 1242 m-tiles per scale
    const int MTB = (MT + 3) / 4;              // 311 blocks of 4 m-tiles

    // 1) weight converts (x_proj padded 48 -> 64 rows)
    {
        long n1 = (long)NS * 1024 * 256, n3 = (long)NS * 256 * 512;
        long n2 = (long)NS * XP_N * 512;
        f32_to_f16_kernel<<<dim3((unsigned)((n1 + 255) / 256)), 256, 0, stream>>>(in_proj_w, wh_in, n1);
        convert_xproj_kernel<<<dim3((unsigned)((n2 + 255) / 256)), 256, 0, stream>>>(x_proj_w, wh_xp);
        f32_to_f16_kernel<<<dim3((unsigned)((n3 + 255) / 256)), 256, 0, stream>>>(out_proj, wh_out, n3);
    }
    // 2) gather x views -> f16 token matrix
    {
        long n = (long)NS * ROWS * D_MODEL;
        gather_x_kernel<<<dim3((unsigned)((n + 255) / 256)), 256, 0, stream>>>(x, xh);
    }
    // 3) in_proj: (ROWS x 256) * (1024 x 256)^T -> xz f16
    wmma_gemm_kernel<<<dim3(MTB, 1024 / 64, NS), dim3(32, 4), 0, stream>>>(
        xh, wh_in, nullptr, xz, 1024, 256, ROWS, ROWS,
        (long)ROWS * 256, (long)1024 * 256, (long)ROWS * 1024);
    // 4) causal conv + SiLU
    {
        long n = (long)NS * ROWS * D_INNER;
        conv_silu_kernel<<<dim3((unsigned)((n + 255) / 256)), 256, 0, stream>>>(xz, conv_w, conv_b, xc);
    }
    // 5) x_proj: (ROWS x 512) * (64 x 512)^T -> x_dbl f32 (cols 48..63 zero)
    wmma_gemm_kernel<<<dim3(MTB, 1, NS), dim3(32, 4), 0, stream>>>(
        xc, wh_xp, xdbl, nullptr, XP_N, 512, ROWS, ROWS,
        (long)ROWS * 512, (long)XP_N * 512, (long)ROWS * XP_N);
    // 6) dt + selective scan + gate -> yh f16 (M padded to 1664)
    scan_kernel<<<dim3(MPAD, NS), dim3(512), 0, stream>>>(
        xc, xz, xdbl, dt_proj_w, dt_proj_b, A_log, Dp, yh);
    // 7) out_proj: (1664 x 512) * (256 x 512)^T -> feats f32 (rows < 1656)
    wmma_gemm_kernel<<<dim3(MPAD / 16 / 4, 256 / 64, NS), dim3(32, 4), 0, stream>>>(
        yh, wh_out, feats, nullptr, 256, 512, MPAD, BN,
        (long)MPAD * 512, (long)256 * 512, (long)BN * 256);
    // 8) scale attention softmax + weighted sum -> out
    attn_out_kernel<<<dim3(BN), dim3(256), 0, stream>>>(feats, attn_w, attn_b, out);
}
